// DisplacementTensors_53772990546083
// MI455X (gfx1250) — compile-verified, matrix operand order checked by reading, not run on hardware
//
#include <hip/hip_runtime.h>
#include <hip/hip_bf16.h>
#include <cmath>

// ---------------------------------------------------------------------------
// DisplacementTensors for MI455X (gfx1250, wave32, WMMA)
//
// Stage 1 (edge_kernel): per (lb, 16-edge tile) wave:
//   - radial Gaussian features (8) per edge
//   - MLP via v_wmma_f32_16x16x32_f16, activations kept TRANSPOSED
//     (channels on M, edges on N) so the f32 C/D tile converts lane-locally
//     into the next layer's f16 B operand (no cross-lane traffic).
//   - 416 global_atomic_add_f32 per edge into an L2-resident accumulator
//     (8*2000*416*4B = 26.6MB << 192MB L2).
// Stage 2 (readout_kernel): fused (B*atoms) x 64 GEMMs with exact-precision
//   v_wmma_f32_16x16x4_f32 (16 K-steps of K=4 over 64 channels).
// ---------------------------------------------------------------------------

typedef __attribute__((ext_vector_type(16))) _Float16 v16h;
typedef __attribute__((ext_vector_type(8)))  float    v8f;
typedef __attribute__((ext_vector_type(2)))  float    v2f;

#define N_LB    8
#define N_E     50000
#define N_ATOMS 2000
#define N_B     4
#define CH_PER  416          // 32 (a) + 96 (v) + 288 (d) floats per (lb, atom)
#define OFF_V   32
#define OFF_D   128
#define OUT_A_OFF 0
#define OUT_V_OFF 512000     // 4*2000*64
#define OUT_D_OFF 2048000    // + 4*2000*64*3

__device__ __forceinline__ v8f wmma_f16(v16h a, v16h b, v8f c) {
    return __builtin_amdgcn_wmma_f32_16x16x32_f16(false, a, false, b, (short)0, c, false, false);
}
__device__ __forceinline__ v8f wmma_f32(v2f a, v2f b, v8f c) {
    return __builtin_amdgcn_wmma_f32_16x16x4_f32(false, a, false, b, (short)0, c, false, false);
}

// A-operand fragment (M=out-channels, K=in-channels) from a row-major weight
// matrix W (O x ld). 16-bit A layout: lane m%16 holds row (mtile*16 + lane&15),
// half h maps to K = (h&7) + 16*(h>>3) + 8*(lane>=16) + kbase.
__device__ __forceinline__ v16h load_w_frag(const float* __restrict__ W, int ld,
                                            int kvalid, int mtile, int kbase, int lane) {
    const float* row = W + (size_t)(mtile * 16 + (lane & 15)) * ld;
    const int khi = (lane >> 4) << 3;
    v16h a;
#pragma unroll
    for (int h = 0; h < 16; ++h) {
        int k = (h & 7) + ((h >> 3) << 4) + khi + kbase;
        float w = (k < kvalid) ? row[k] : 0.0f;
        a[h] = (_Float16)w;
    }
    return a;
}

// C-init fragment broadcasting bias over the N (edge) dimension:
// C[m][n] = bias[m];  f32 C tile: vgpr j, lane l -> M = mtile*16 + j + 8*(l>=16).
__device__ __forceinline__ v8f bias_frag(const float* __restrict__ b, int mtile, int lane) {
    const int base = mtile * 16 + ((lane >> 4) << 3);
    v8f c;
#pragma unroll
    for (int j = 0; j < 8; ++j) c[j] = b[base + j];
    return c;
}

// Convert a 32xN f32 activation (two 16x16 C tiles, rows 0-15 and 16-31) into
// the next layer's f16 B operand. With B mirroring the documented 16-bit A
// layout, half h needs K = (h&7) + 16*(h>>3) + 8*(lane>=16), which is exactly
// C-tile vgpr (h&7) of tile (h>>3) in the SAME lane: pure per-lane converts.
__device__ __forceinline__ v16h pack_b(v8f t0, v8f t1) {
    v16h r;
#pragma unroll
    for (int h = 0; h < 8; ++h) {
        r[h]     = (_Float16)t0[h];
        r[h + 8] = (_Float16)t1[h];
    }
    return r;
}

__device__ __forceinline__ v8f leaky8(v8f x) {
#pragma unroll
    for (int j = 0; j < 8; ++j) x[j] = fmaxf(x[j], 0.1f * x[j]);
    return x;
}

// ---------------------------------------------------------------------------
__global__ __launch_bounds__(256) void zero_kernel(float4* __restrict__ p, int n4) {
    int i = blockIdx.x * 256 + threadIdx.x;
    if (i < n4) p[i] = make_float4(0.f, 0.f, 0.f, 0.f);
}

// ---------------------------------------------------------------------------
__global__ __launch_bounds__(256) void edge_kernel(
    const float* __restrict__ r_ij, const int* __restrict__ src,
    const float* __restrict__ w_in, const float* __restrict__ b_in,
    const float* __restrict__ w_dir,
    const float* __restrict__ w1, const float* __restrict__ b1,
    const float* __restrict__ w2, const float* __restrict__ b2,
    const float* __restrict__ w3, const float* __restrict__ b3,
    float* __restrict__ acc)
{
    const int lane = threadIdx.x & 31;
    const int wave = threadIdx.x >> 5;
    const int task = blockIdx.x * 8 + wave;       // 3125 blocks * 8 = 25000 tasks
    const int lb   = task / 3125;                 // 8 lb slices
    const int e0   = (task - lb * 3125) * 16;     // 3125 * 16 = 50000 edges, exact
    const int el   = lane & 15;
    const int e    = e0 + el;

    const float* rp = r_ij + ((size_t)lb * N_E + e) * 3;
    const float rx = rp[0], ry = rp[1], rz = rp[2];
    const float r  = sqrtf(rx * rx + ry * ry + rz * rz);

    // --- radial features -> B operand of layer 0 (K padded 8 -> 32) -------
    // B half h in lane l is K = (h&7)+16*(h>>3)+8*(l>=16): only lanes 0-15,
    // halves 0..7 carry live features (K=0..7); everything else is zero-pad.
    const float x = r * (1.0f / 6.0f);            // r / R0
    v16h bfeat;
#pragma unroll
    for (int h = 0; h < 16; ++h) bfeat[h] = (_Float16)0.0f;
    if (lane < 16) {
#pragma unroll
        for (int i = 0; i < 8; ++i) {
            float d = x - (float)i * (1.0f / 7.0f);
            bfeat[i] = (_Float16)expf(-32.0f * d * d);   // 0.5*((x-c)*8)^2
        }
    }

    // --- h^T = w_in @ feat^T + b_in   (32 x 16) ---------------------------
    v8f H0 = wmma_f16(load_w_frag(w_in, 8, 8, 0, 0, lane), bfeat, bias_frag(b_in, 0, lane));
    v8f H1 = wmma_f16(load_w_frag(w_in, 8, 8, 1, 0, lane), bfeat, bias_frag(b_in, 1, lane));
    v16h bH = pack_b(H0, H1);

    // --- hh1 = leaky(w1 @ h + b1)     (64 x 16), K=32 ---------------------
    v8f G[4];
#pragma unroll
    for (int t = 0; t < 4; ++t) {
        v8f c = bias_frag(b1, t, lane);
        c = wmma_f16(load_w_frag(w1, 32, 32, t, 0, lane), bH, c);
        G[t] = leaky8(c);
    }
    v16h bG0 = pack_b(G[0], G[1]);
    v16h bG1 = pack_b(G[2], G[3]);

    // --- hh2 = leaky(w2 @ hh1 + b2)   (64 x 16), K=64 ---------------------
    v8f U[4];
#pragma unroll
    for (int t = 0; t < 4; ++t) {
        v8f c = bias_frag(b2, t, lane);
        c = wmma_f16(load_w_frag(w2, 64, 64, t, 0,  lane), bG0, c);
        c = wmma_f16(load_w_frag(w2, 64, 64, t, 32, lane), bG1, c);
        U[t] = leaky8(c);
    }
    v16h bU0 = pack_b(U[0], U[1]);
    v16h bU1 = pack_b(U[2], U[3]);

    // --- rad = w_dir @ h + w3 @ hh2 + b3   (32 x 16) ----------------------
    v8f R[2];
#pragma unroll
    for (int t = 0; t < 2; ++t) {
        v8f c = bias_frag(b3, t, lane);
        c = wmma_f16(load_w_frag(w3, 64, 64, t, 0,  lane), bU0, c);
        c = wmma_f16(load_w_frag(w3, 64, 64, t, 32, lane), bU1, c);
        c = wmma_f16(load_w_frag(w_dir, 32, 32, t, 0, lane), bH, c);
        R[t] = c;
    }

    // --- rs = tens_sigmoid(r_ij * 7/R0) -----------------------------------
    const float nn = r * (7.0f / 6.0f);
    const float sc = tanhf(nn) / fmaxf(nn, 1e-6f);
    const float s0 = rx * (7.0f / 6.0f) * sc;
    const float s1 = ry * (7.0f / 6.0f) * sc;
    const float s2 = rz * (7.0f / 6.0f) * sc;
    const float sv[3] = { s0, s1, s2 };
    const float ss[9] = { s0*s0, s0*s1, s0*s2,
                          s1*s0, s1*s1, s1*s2,
                          s2*s0, s2*s1, s2*s2 };

    // --- scatter: rad[c][e] * {1, rs, rs (x) rs} into per-atom accumulator.
    // Lane l holds channels {16t + j + 8*(l>=16)} of edge l%16.
    const int atom = src[e];
    float* base = acc + ((size_t)lb * N_ATOMS + atom) * CH_PER;
    const int chi = (lane >> 4) << 3;
#pragma unroll
    for (int t = 0; t < 2; ++t) {
#pragma unroll
        for (int j = 0; j < 8; ++j) {
            const int   c = t * 16 + chi + j;
            const float v = R[t][j];
            atomicAdd(base + c, v);
            float* pv = base + OFF_V + c * 3;
            for (int q = 0; q < 3; ++q) atomicAdd(pv + q, v * sv[q]);
            float* pd = base + OFF_D + c * 9;
            for (int q = 0; q < 9; ++q) atomicAdd(pd + q, v * ss[q]);
        }
    }
}

// ---------------------------------------------------------------------------
// Readout: rows = b*2000+n (8000 -> 500 M-tiles of 16). Fused channel
// c' = l*32 + c pulls from lb = l*4 + b slice of the accumulator.
// Per wave: one 16x16 output tile, 16 x wmma_f32_16x16x4_f32 K-steps.
// 34 output tiles per M-tile: a (4x N16), v (3 comps x 4), d (9 comps x 2).
__global__ __launch_bounds__(256) void readout_kernel(
    const float* __restrict__ acc,
    const float* __restrict__ w_a, const float* __restrict__ w_v,
    const float* __restrict__ w_d, float* __restrict__ out)
{
    const int lane  = threadIdx.x & 31;
    const int wave  = threadIdx.x >> 5;
    const int task  = blockIdx.x * 8 + wave;      // 2125 * 8 = 17000 tasks, exact
    const int mtile = task / 34;
    const int ot    = task - mtile * 34;

    int type, otile, comp;
    const float* W;
    if (ot < 4)       { type = 0; otile = ot;                  comp = 0;               W = w_a; }
    else if (ot < 16) { int i = ot - 4;  type = 1; otile = i & 3; comp = i >> 2;       W = w_v; }
    else              { int i = ot - 16; type = 2; otile = i & 1; comp = i >> 1;       W = w_d; }

    const int mrow = mtile * 16 + (lane & 15);    // A-side row for this lane
    const int bA   = mrow / N_ATOMS;
    const int nA   = mrow - bA * N_ATOMS;
    const int o    = otile * 16 + (lane & 15);    // B-side output column
    const int k2   = (lane >> 4) << 1;            // 32-bit A/B: lanes>=16 hold K=2,3

    v8f cacc;
#pragma unroll
    for (int j = 0; j < 8; ++j) cacc[j] = 0.0f;

#pragma unroll
    for (int kb = 0; kb < 64; kb += 4) {
        v2f af, bf;
#pragma unroll
        for (int u = 0; u < 2; ++u) {
            const int k = kb + k2 + u;            // fused channel c' = l*32 + c
            const int l = k >> 5, c = k & 31;
            const size_t row = ((size_t)(l * N_B + bA) * N_ATOMS + nA) * CH_PER;
            const int off = (type == 0) ? c
                          : (type == 1) ? (OFF_V + c * 3 + comp)
                                        : (OFF_D + c * 9 + comp);
            af[u] = acc[row + off];
            bf[u] = W[(size_t)o * 64 + k];
        }
        cacc = wmma_f32(af, bf, cacc);
    }

    // f32 D tile: vgpr j, lane l -> M = j + 8*(l>=16), N = lane&15
    const int mhi = (lane >> 4) << 3;
#pragma unroll
    for (int j = 0; j < 8; ++j) {
        const int row = mtile * 16 + mhi + j;
        const int b   = row / N_ATOMS;
        const int n   = row - b * N_ATOMS;
        size_t oi;
        if (type == 0)      oi = OUT_A_OFF + ((size_t)(b * N_ATOMS + n) * 64 + o);
        else if (type == 1) oi = OUT_V_OFF + (((size_t)(b * N_ATOMS + n) * 64 + o) * 3 + comp);
        else                oi = OUT_D_OFF + (((size_t)(b * N_ATOMS + n) * 32 + o) * 9 + comp);
        out[oi] = cacc[j];
    }
}

// ---------------------------------------------------------------------------
extern "C" void kernel_launch(void* const* d_in, const int* in_sizes, int n_in,
                              void* d_out, int out_size, void* d_ws, size_t ws_size,
                              hipStream_t stream)
{
    const float* r_ij  = (const float*)d_in[0];
    const int*   src   = (const int*)  d_in[1];
    const float* w_in  = (const float*)d_in[2];
    const float* b_in  = (const float*)d_in[3];
    const float* w_dir = (const float*)d_in[4];
    const float* w1    = (const float*)d_in[5];
    const float* b1    = (const float*)d_in[6];
    const float* w2    = (const float*)d_in[7];
    const float* b2    = (const float*)d_in[8];
    const float* w3    = (const float*)d_in[9];
    const float* b3    = (const float*)d_in[10];
    const float* w_a   = (const float*)d_in[11];
    const float* w_v   = (const float*)d_in[12];
    const float* w_d   = (const float*)d_in[13];

    float* out = (float*)d_out;
    float* acc = (float*)d_ws;                        // 8*2000*416 f32 = 26.6 MB

    const int accN  = N_LB * N_ATOMS * CH_PER;        // 6,656,000 floats
    const int accN4 = accN / 4;                       // 1,664,000 float4
    zero_kernel<<<accN4 / 256, 256, 0, stream>>>((float4*)acc, accN4);
    edge_kernel<<<3125, 256, 0, stream>>>(r_ij, src, w_in, b_in, w_dir,
                                          w1, b1, w2, b2, w3, b3, acc);
    readout_kernel<<<2125, 256, 0, stream>>>(acc, w_a, w_v, w_d, out);
}
